// Gnn_5188320494328
// MI455X (gfx1250) — matrix operand order repaired and checked by reading, not compiled, not run on hardware
//
#include <hip/hip_runtime.h>

#define B_DIM 4096
#define C_DIM 62
#define F_INN 256
#define F_BNN 64
#define F_OUTN 64
#define TOPKN 10

typedef __attribute__((ext_vector_type(16))) __bf16 v16bf;
typedef __attribute__((ext_vector_type(8)))  __bf16 v8bf;
typedef __attribute__((ext_vector_type(8)))  float  v8f;

// Native f32 -> bf16 conversion (gfx1250 has hardware cvt; clang lowers the cast)
__device__ __forceinline__ __bf16 f2bf(float f) { return (__bf16)f; }

// Fast reciprocal via v_rcp_f32
__device__ __forceinline__ float fast_rcp(float f) { return __builtin_amdgcn_rcpf(f); }

struct WaveMem {
  float  S[64][66];       // S scores, then masked adj (f32). stride 66 => bank-spread
  __bf16 xaAdj[64][72];   // xa (phases 1-2), then normalized adj bf16 (phases 3-4)
  __bf16 xgT[64][72];     // xg transposed: [feature][node]
  float  d[64];           // deg^-1/2 per row
};

struct BlockMem {
  __bf16 WtBN[64][264];   // W_bn^T bf16, padded stride
  __bf16 WtG [64][264];   // W_g^T  bf16
  WaveMem wm[4];
};

__device__ __forceinline__ v16bf comb(v8bf lo, v8bf hi) {
  return __builtin_shufflevector(lo, hi, 0,1,2,3,4,5,6,7,8,9,10,11,12,13,14,15);
}

// A-tile (16x32 bf16) from LDS row-major: lane<16 -> K {k0..k0+7, k0+16..k0+23}
__device__ __forceinline__ v16bf loadA_lds(const __bf16* rowptr, int k0) {
  v8bf lo = *(const v8bf*)(rowptr + k0);
  v8bf hi = *(const v8bf*)(rowptr + k0 + 16);
  return comb(lo, hi);
}

// B-tile (32x16 bf16): lane holds fixed N, K {kb..kb+7, kb+16..kb+23}
__device__ __forceinline__ v16bf loadB_lds(const __bf16* rowptr, int kb) {
  v8bf lo = *(const v8bf*)(rowptr + kb);
  v8bf hi = *(const v8bf*)(rowptr + kb + 16);
  return comb(lo, hi);
}

__device__ __forceinline__ v8f wmma_bf16(v16bf a, v16bf b, v8f c) {
  return __builtin_amdgcn_wmma_f32_16x16x32_bf16(false, a, false, b, (short)0, c,
                                                 false, false);
}

__global__ __launch_bounds__(128, 1) void gnn_fused_kernel(
    const float* __restrict__ x, const float* __restrict__ Wbn,
    const float* __restrict__ bbn, const float* __restrict__ Wg,
    const float* __restrict__ bg, float* __restrict__ out1,
    float* __restrict__ out2)
{
  __shared__ BlockMem sm;
  const int tid  = threadIdx.x;
  const int wave = tid >> 5;
  const int lane = tid & 31;
  const int b    = blockIdx.x * 4 + wave;
  WaveMem& wm = sm.wm[wave];

  // ---- Phase 0: cooperative load of W_bn^T / W_g^T into LDS as bf16 ----
  for (int idx = tid; idx < F_INN * F_BNN; idx += 128) {
    int k = idx >> 6, n = idx & 63;          // W row-major [k][n]
    sm.WtBN[n][k] = f2bf(Wbn[idx]);
    sm.WtG [n][k] = f2bf(Wg[idx]);
  }
  __syncthreads();

  const int lhi = lane >> 4;   // 0 or 1 (half-wave)
  const int llo = lane & 15;

  // ---- Phase 1: xa = tanh(x@Wbn + bbn) ; xg = x@Wg  (shared A tiles) ----
  for (int mt = 0; mt < 4; ++mt) {
    v8f accBN[4] = {};
    v8f accG [4] = {};
    const int arow = mt * 16 + llo;
    const bool valid = arow < C_DIM;
    #pragma unroll
    for (int kt = 0; kt < 8; ++kt) {
      const int ka = kt * 32 + lhi * 8;
      v16bf a;
      if (valid) {
        const float* xp = x + ((size_t)b * C_DIM + arow) * F_INN + ka;
        // pull the next k-slice toward the WGP while this one converts
        __builtin_prefetch(xp + 32, 0, 3);
        float4 f0 = ((const float4*)xp)[0];
        float4 f1 = ((const float4*)xp)[1];
        float4 f2 = ((const float4*)(xp + 16))[0];
        float4 f3 = ((const float4*)(xp + 16))[1];
        a[0]=f2bf(f0.x);  a[1]=f2bf(f0.y);  a[2]=f2bf(f0.z);  a[3]=f2bf(f0.w);
        a[4]=f2bf(f1.x);  a[5]=f2bf(f1.y);  a[6]=f2bf(f1.z);  a[7]=f2bf(f1.w);
        a[8]=f2bf(f2.x);  a[9]=f2bf(f2.y);  a[10]=f2bf(f2.z); a[11]=f2bf(f2.w);
        a[12]=f2bf(f3.x); a[13]=f2bf(f3.y); a[14]=f2bf(f3.z); a[15]=f2bf(f3.w);
      } else {
        const __bf16 z = f2bf(0.f);
        #pragma unroll
        for (int i = 0; i < 16; ++i) a[i] = z;
      }
      const int kb = kt * 32 + lhi * 8;
      #pragma unroll
      for (int nt = 0; nt < 4; ++nt) {
        accBN[nt] = wmma_bf16(a, loadB_lds(&sm.WtBN[nt*16 + llo][0], kb), accBN[nt]);
        accG [nt] = wmma_bf16(a, loadB_lds(&sm.WtG [nt*16 + llo][0], kb), accG [nt]);
      }
    }
    const int mbase = mt * 16 + lhi * 8;
    #pragma unroll
    for (int nt = 0; nt < 4; ++nt) {
      const int n = nt * 16 + llo;
      const float bias = bbn[n];
      v8bf pk;
      #pragma unroll
      for (int v = 0; v < 8; ++v) {
        float t = accBN[nt][v] + bias;
        t = fminf(fmaxf(t, -15.f), 15.f);
        float e = __expf(2.f * t);                       // tanh = (e-1)/(e+1)
        wm.xaAdj[mbase + v][n] = f2bf((e - 1.f) * fast_rcp(e + 1.f));
        pk[v] = f2bf(accG[nt][v]);                       // store xg transposed
      }
      *(v8bf*)&wm.xgT[n][mbase] = pk;
    }
  }
  __syncthreads();

  // ---- Phase 2: S = xa @ xa^T (K = F_BN = 64) ----
  for (int mt = 0; mt < 4; ++mt) {
    v8f acc[4] = {};
    #pragma unroll
    for (int kt = 0; kt < 2; ++kt) {
      v16bf a = loadA_lds(&wm.xaAdj[mt*16 + llo][0], kt*32 + lhi*8);
      const int kb = kt*32 + lhi*8;
      #pragma unroll
      for (int nt = 0; nt < 4; ++nt)
        acc[nt] = wmma_bf16(a, loadB_lds(&wm.xaAdj[nt*16 + llo][0], kb), acc[nt]);
    }
    const int mbase = mt*16 + lhi*8;
    #pragma unroll
    for (int nt = 0; nt < 4; ++nt) {
      const int n = nt*16 + llo;
      #pragma unroll
      for (int v = 0; v < 8; ++v) wm.S[mbase + v][n] = acc[nt][v];
    }
  }
  __syncthreads();

  // ---- Phase 3a: per-row softmax, top-10 mask, diag=1, degree ----
  #pragma unroll
  for (int half = 0; half < 2; ++half) {
    const int r = lane + 32 * half;
    if (r < C_DIM) {
      float* Srow = wm.S[r];
      float mx = -3.4e38f;
      for (int j = 0; j < C_DIM; ++j) mx = fmaxf(mx, Srow[j]);
      float sum = 0.f;
      for (int j = 0; j < C_DIM; ++j) { float e = __expf(Srow[j] - mx); Srow[j] = e; sum += e; }
      const float inv = fast_rcp(sum);
      unsigned selLo = 0, selHi = 0;
      for (int t = 0; t < TOPKN; ++t) {
        float bv = -1.f; int bi = 0;
        for (int j = 0; j < C_DIM; ++j) {
          unsigned sbit = (j < 32) ? (selLo >> j) : (selHi >> (j - 32));
          float vv = Srow[j];
          if (!(sbit & 1u) && vv > bv) { bv = vv; bi = j; }
        }
        if (bi < 32) selLo |= 1u << bi; else selHi |= 1u << (bi - 32);
      }
      float deg = 0.f;
      for (int j = 0; j < C_DIM; ++j) {
        unsigned sbit = (j < 32) ? (selLo >> j) : (selHi >> (j - 32));
        float vv = (sbit & 1u) ? Srow[j] * inv : 0.f;
        if (j == r) vv = 1.f;                     // diag set to 1 after masking
        Srow[j] = vv;
        deg += vv;
      }
      wm.d[r] = rsqrtf(fmaxf(deg, 1.f));
    }
  }
  if (lane < 2) wm.d[62 + lane] = 0.f;
  __syncthreads();

  // ---- Phase 3b: adj_out -> global (coalesced); build normalized bf16 adj ----
  {
    float* o2 = out2 + (size_t)b * C_DIM * C_DIM;
    for (int idx = lane; idx < C_DIM * C_DIM; idx += 32)
      o2[idx] = wm.S[idx / C_DIM][idx % C_DIM];
    #pragma unroll
    for (int half = 0; half < 2; ++half) {
      const int r = lane + 32 * half;
      if (r < C_DIM) {
        const float dr = wm.d[r];
        for (int j = 0; j < C_DIM; ++j)
          wm.xaAdj[r][j] = f2bf(wm.S[r][j] * dr * wm.d[j]);
        wm.xaAdj[r][62] = f2bf(0.f);
        wm.xaAdj[r][63] = f2bf(0.f);
      }
    }
    for (int idx = lane; idx < 64; idx += 32) {
      wm.xaAdj[62][idx] = f2bf(0.f);
      wm.xaAdj[63][idx] = f2bf(0.f);
    }
  }
  __syncthreads();

  // ---- Phase 4: out = relu(adj_norm @ xg + bg) ----
  for (int mt = 0; mt < 4; ++mt) {
    v8f acc[4] = {};
    #pragma unroll
    for (int kt = 0; kt < 2; ++kt) {
      v16bf a = loadA_lds(&wm.xaAdj[mt*16 + llo][0], kt*32 + lhi*8);
      const int kb = kt*32 + lhi*8;
      #pragma unroll
      for (int nt = 0; nt < 4; ++nt)
        acc[nt] = wmma_bf16(a, loadB_lds(&wm.xgT[nt*16 + llo][0], kb), acc[nt]);
    }
    const int mbase = mt*16 + lhi*8;
    #pragma unroll
    for (int nt = 0; nt < 4; ++nt) {
      const int n = nt*16 + llo;
      const float bias = bg[n];
      #pragma unroll
      for (int v = 0; v < 8; ++v) {
        const int mm = mbase + v;
        if (mm < C_DIM)
          out1[((size_t)b * C_DIM + mm) * F_OUTN + n] = fmaxf(acc[nt][v] + bias, 0.f);
      }
    }
  }
}

extern "C" void kernel_launch(void* const* d_in, const int* in_sizes, int n_in,
                              void* d_out, int out_size, void* d_ws, size_t ws_size,
                              hipStream_t stream) {
  const float* x   = (const float*)d_in[0];
  const float* Wbn = (const float*)d_in[1];
  const float* bbn = (const float*)d_in[2];
  const float* Wg  = (const float*)d_in[3];
  const float* bg  = (const float*)d_in[4];
  float* out1 = (float*)d_out;                                  // [B, C, F_OUT]
  float* out2 = out1 + (size_t)B_DIM * C_DIM * F_OUTN;          // [B, C, C]
  dim3 grid(B_DIM / 4), block(128);
  gnn_fused_kernel<<<grid, block, 0, stream>>>(x, Wbn, bbn, Wg, bg, out1, out2);
}
